// QuantizedDecoder_45827301048795
// MI455X (gfx1250) — matrix operand order repaired and checked by reading, not compiled, not run on hardware
//
#include <hip/hip_runtime.h>

// ---------------------------------------------------------------------------
// QuantizedDecoder fused pipeline for MI455X (gfx1250, wave32, WMMA bf16)
//   B=131072, IN_DIM=256, HID=512, LAT=256, K(codebook)=1024
//   32-row M-tile per wave: each B fragment feeds 2 WMMAs (halves L2 traffic)
// ---------------------------------------------------------------------------

#define B_ROWS 131072
#define IN_DIM 256
#define HID    512
#define LAT    256
#define KCB    1024

typedef __bf16 bf16_t;
typedef bf16_t bfv16 __attribute__((ext_vector_type(16)));
typedef bf16_t bfv8  __attribute__((ext_vector_type(8)));
typedef float  fv8   __attribute__((ext_vector_type(8)));

__device__ __forceinline__ unsigned short f2bf(float f) {
  unsigned int u = __builtin_bit_cast(unsigned int, f);
  unsigned int r = u + 0x7fffu + ((u >> 16) & 1u);   // round-to-nearest-even
  return (unsigned short)(r >> 16);
}
__device__ __forceinline__ float bf2f(unsigned short h) {
  return __builtin_bit_cast(float, (unsigned int)h << 16);
}

// --------------------------- prep kernels ----------------------------------

__global__ void convert_f32_bf16(const float* __restrict__ src,
                                 unsigned short* __restrict__ dst, int n) {
  int i = blockIdx.x * blockDim.x + threadIdx.x;
  int stride = gridDim.x * blockDim.x;
  for (; i < n; i += stride) dst[i] = f2bf(src[i]);
}

// codebook (LAT x KCB, row-major) -> cbT (KCB x LAT bf16) + cbT32 (KCB x LAT f32)
__global__ void prep_codebook(const float* __restrict__ cb,
                              unsigned short* __restrict__ cbT,
                              float* __restrict__ cbT32) {
  int i = blockIdx.x * blockDim.x + threadIdx.x;
  int total = LAT * KCB;
  int stride = gridDim.x * blockDim.x;
  for (; i < total; i += stride) {
    int l = i / KCB, j = i % KCB;
    float v = cb[i];
    cbT[(size_t)j * LAT + l]   = f2bf(v);
    cbT32[(size_t)j * LAT + l] = v;
  }
}

__global__ void cnorm_kernel(const float* __restrict__ cb, float* __restrict__ cnorm) {
  int j = blockIdx.x * blockDim.x + threadIdx.x;
  if (j >= KCB) return;
  float s = 0.f;
  for (int l = 0; l < LAT; ++l) { float v = cb[(size_t)l * KCB + j]; s += v * v; }
  cnorm[j] = s;
}

__global__ void zero_kernel(float* p) {
  if (blockIdx.x == 0 && threadIdx.x == 0) p[0] = 0.f;
}

// --------------------------- WMMA GEMM -------------------------------------
// out[m,n] = act( A[m,:] . W[n,:] + bias[n] )   A: M x K bf16 (lda), W: N x K bf16
// one wave per block: 32 rows x 128 cols; every B fragment feeds 2 WMMAs.
template<int RELU, int STOREF32>
__global__ __launch_bounds__(32)
void gemm_wmma_bf16(const unsigned short* __restrict__ Au, int lda,
                    const unsigned short* __restrict__ Wu, int K,
                    const float* __restrict__ bias,
                    unsigned short* __restrict__ outb, int ldo,
                    float* __restrict__ outf) {
  const int lane = threadIdx.x;
  const int half = lane >> 4;
  const int l16  = lane & 15;
  const int m0 = blockIdx.x * 32;
  const int n0 = blockIdx.y * 128;
  const bf16_t* A = (const bf16_t*)Au;
  const bf16_t* W = (const bf16_t*)Wu;

  fv8 acc[2][8] = {};

  // A-fragment: lane l16 = row, half selects K sub-chunks {0..15}/{16..31}
  const bf16_t* arow0 = A + (size_t)(m0 + l16) * lda + half * 16;
  const bf16_t* arow1 = A + (size_t)(m0 + 16 + l16) * lda + half * 16;
  for (int k0 = 0; k0 < K; k0 += 32) {
    union { bfv16 v; bfv8 h[2]; } a0, a1;
    a0.h[0] = *(const bfv8*)(arow0 + k0);
    a0.h[1] = *(const bfv8*)(arow0 + k0 + 8);
    a1.h[0] = *(const bfv8*)(arow1 + k0);
    a1.h[1] = *(const bfv8*)(arow1 + k0 + 8);
#pragma unroll
    for (int t = 0; t < 8; ++t) {
      // B-fragment: lane = column n0+t*16+l16; 16 contiguous K elems (by half)
      bfv16 b = *(const bfv16*)(W + (size_t)(n0 + t * 16 + l16) * K + k0 + half * 16);
      acc[0][t] = __builtin_amdgcn_wmma_f32_16x16x32_bf16(
          false, a0.v, false, b, (short)0, acc[0][t], false, false);
      acc[1][t] = __builtin_amdgcn_wmma_f32_16x16x32_bf16(
          false, a1.v, false, b, (short)0, acc[1][t], false, false);
    }
  }
#pragma unroll
  for (int mt = 0; mt < 2; ++mt) {
#pragma unroll
    for (int t = 0; t < 8; ++t) {
      int col = n0 + t * 16 + l16;
      float bv = bias[col];
#pragma unroll
      for (int r = 0; r < 8; ++r) {
        size_t row = (size_t)(m0 + mt * 16 + r + 8 * half);
        float v = acc[mt][t][r] + bv;
        if (RELU) v = v > 0.f ? v : 0.f;
        outb[row * ldo + col] = f2bf(v);
        if (STOREF32) outf[row * ldo + col] = v;
      }
    }
  }
}

// --------------------------- fout==1 heads ---------------------------------
__global__ __launch_bounds__(256)
void head_dot(const unsigned short* __restrict__ H,   // B x HID bf16
              const unsigned short* __restrict__ Wv,  // HID bf16
              const float* __restrict__ bias,         // [1]
              float* __restrict__ out, int relu) {
  int wave = (int)((blockIdx.x * blockDim.x + threadIdx.x) >> 5);
  int lane = threadIdx.x & 31;
  if (wave >= B_ROWS) return;
  const unsigned short* row = H + (size_t)wave * HID;
  float s = 0.f;
  for (int k = lane; k < HID; k += 32) s += bf2f(row[k]) * bf2f(Wv[k]);
#pragma unroll
  for (int off = 16; off >= 1; off >>= 1) s += __shfl_down(s, off, 32);
  if (lane == 0) {
    float v = s + bias[0];
    if (relu) v = v > 0.f ? v : 0.f;
    out[wave] = v;
  }
}

// --------------------------- VQ: argmin + gather + loss --------------------
// scores s[m,j] = |c_j|^2 - 2 * zl[m,:].c_j  (|zl|^2 constant per row -> drop)
// 32 rows per wave: full 32x256 A tile held in registers across all 64 code tiles,
// so every codebook fragment load feeds 2 WMMAs.
__global__ __launch_bounds__(32)
void vq_argmin(const unsigned short* __restrict__ zlb,  // B x LAT bf16
               const unsigned short* __restrict__ cbT,  // KCB x LAT bf16
               const float* __restrict__ cbT32,         // KCB x LAT f32
               const float* __restrict__ cnorm,         // KCB
               float* __restrict__ zq,                  // B x LAT (holds zl f32, rewritten to z_q)
               float* __restrict__ ids,
               float* __restrict__ loss) {
  __shared__ int sidx[32];
  const int lane = threadIdx.x;
  const int half = lane >> 4;
  const int l16  = lane & 15;
  const int m0 = blockIdx.x * 32;
  const bf16_t* Z = (const bf16_t*)zlb;
  const bf16_t* C = (const bf16_t*)cbT;

  // preload the full 32x256 A tile (reused for all 64 codebook N-tiles)
  union { bfv16 v; bfv8 h[2]; } a[2][8];
  const bf16_t* arow0 = Z + (size_t)(m0 + l16) * LAT + half * 16;
  const bf16_t* arow1 = Z + (size_t)(m0 + 16 + l16) * LAT + half * 16;
#pragma unroll
  for (int kk = 0; kk < 8; ++kk) {
    a[0][kk].h[0] = *(const bfv8*)(arow0 + kk * 32);
    a[0][kk].h[1] = *(const bfv8*)(arow0 + kk * 32 + 8);
    a[1][kk].h[0] = *(const bfv8*)(arow1 + kk * 32);
    a[1][kk].h[1] = *(const bfv8*)(arow1 + kk * 32 + 8);
  }

  float best[2][8];
  int   bidx[2][8];
#pragma unroll
  for (int mt = 0; mt < 2; ++mt)
#pragma unroll
    for (int r = 0; r < 8; ++r) { best[mt][r] = 3.4e38f; bidx[mt][r] = 0; }

  for (int jt = 0; jt < KCB / 16; ++jt) {
    fv8 acc0 = {}, acc1 = {};
    const bf16_t* brow = C + (size_t)(jt * 16 + l16) * LAT + half * 16;
#pragma unroll
    for (int kk = 0; kk < 8; ++kk) {
      bfv16 b = *(const bfv16*)(brow + kk * 32);
      acc0 = __builtin_amdgcn_wmma_f32_16x16x32_bf16(
          false, a[0][kk].v, false, b, (short)0, acc0, false, false);
      acc1 = __builtin_amdgcn_wmma_f32_16x16x32_bf16(
          false, a[1][kk].v, false, b, (short)0, acc1, false, false);
    }
    int col = jt * 16 + l16;
    float cn = cnorm[col];
#pragma unroll
    for (int r = 0; r < 8; ++r) {
      float s0 = cn - 2.f * acc0[r];
      float s1 = cn - 2.f * acc1[r];
      if (s0 < best[0][r]) { best[0][r] = s0; bidx[0][r] = col; }
      if (s1 < best[1][r]) { best[1][r] = s1; bidx[1][r] = col; }
    }
  }

  // reduce min/argmin across the 16 lanes of each half (xor<16 stays in-group)
#pragma unroll
  for (int mt = 0; mt < 2; ++mt) {
#pragma unroll
    for (int r = 0; r < 8; ++r) {
#pragma unroll
      for (int off = 8; off >= 1; off >>= 1) {
        float ob = __shfl_xor(best[mt][r], off, 32);
        int   oi = __shfl_xor(bidx[mt][r], off, 32);
        if (ob < best[mt][r] || (ob == best[mt][r] && oi < bidx[mt][r])) {
          best[mt][r] = ob; bidx[mt][r] = oi;
        }
      }
      if (l16 == r) {
        int row = m0 + mt * 16 + r + 8 * half;
        ids[row] = (float)bidx[mt][r];
        sidx[mt * 16 + r + 8 * half] = bidx[mt][r];
      }
    }
  }
  __syncthreads();

  // gather exact f32 code vectors, overwrite zl with z_q, accumulate loss
  float lsum = 0.f;
  for (int rr = 0; rr < 32; ++rr) {
    int row = m0 + rr;
    const float* cq = cbT32 + (size_t)sidx[rr] * LAT;
    float* zrow = zq + (size_t)row * LAT;
#pragma unroll
    for (int i = 0; i < LAT / 32; ++i) {
      int l = lane + i * 32;
      float q = cq[l];
      float d = q - zrow[l];
      lsum += d * d;
      zrow[l] = q;           // z_q_st forward value == z_q
    }
  }
#pragma unroll
  for (int off = 16; off >= 1; off >>= 1) lsum += __shfl_down(lsum, off, 32);
  if (lane == 0) atomicAdd(loss, lsum * (1.0f / ((float)B_ROWS * (float)LAT)));
}

// --------------------------- launch ----------------------------------------

extern "C" void kernel_launch(void* const* d_in, const int* in_sizes, int n_in,
                              void* d_out, int out_size, void* d_ws, size_t ws_size,
                              hipStream_t stream) {
  (void)in_sizes; (void)n_in; (void)out_size; (void)ws_size;

  const float* z        = (const float*)d_in[0];
  const float* dec_W0   = (const float*)d_in[1];
  const float* dec_b0   = (const float*)d_in[2];
  const float* dec_W1   = (const float*)d_in[3];
  const float* dec_b1   = (const float*)d_in[4];
  const float* dec_Wout = (const float*)d_in[5];
  const float* dec_bout = (const float*)d_in[6];
  const float* sc_W0    = (const float*)d_in[7];
  const float* sc_b0    = (const float*)d_in[8];
  const float* sc_W1    = (const float*)d_in[9];
  const float* sc_b1    = (const float*)d_in[10];
  const float* sc_Wout  = (const float*)d_in[11];
  const float* sc_bout  = (const float*)d_in[12];
  const float* rs_W0    = (const float*)d_in[13];
  const float* rs_b0    = (const float*)d_in[14];
  const float* rs_W1    = (const float*)d_in[15];
  const float* rs_b1    = (const float*)d_in[16];
  const float* rs_Wout  = (const float*)d_in[17];
  const float* rs_bout  = (const float*)d_in[18];
  const float* codebook = (const float*)d_in[19];

  char* ws = (char*)d_ws;
  size_t off = 0;
  auto alloc = [&](size_t bytes) -> char* {
    char* p = ws + off; off += (bytes + 255) & ~(size_t)255; return p;
  };
  unsigned short* zbf   = (unsigned short*)alloc((size_t)B_ROWS * IN_DIM * 2);
  unsigned short* hA    = (unsigned short*)alloc((size_t)B_ROWS * HID * 2);
  unsigned short* hB    = (unsigned short*)alloc((size_t)B_ROWS * HID * 2);
  unsigned short* zlb   = (unsigned short*)alloc((size_t)B_ROWS * LAT * 2);
  unsigned short* wdec0 = (unsigned short*)alloc((size_t)HID * IN_DIM * 2);
  unsigned short* wdec1 = (unsigned short*)alloc((size_t)HID * HID * 2);
  unsigned short* wdecO = (unsigned short*)alloc((size_t)LAT * HID * 2);
  unsigned short* wsc0  = (unsigned short*)alloc((size_t)HID * IN_DIM * 2);
  unsigned short* wsc1  = (unsigned short*)alloc((size_t)HID * HID * 2);
  unsigned short* wscO  = (unsigned short*)alloc((size_t)HID * 2);
  unsigned short* wrs0  = (unsigned short*)alloc((size_t)HID * IN_DIM * 2);
  unsigned short* wrs1  = (unsigned short*)alloc((size_t)HID * HID * 2);
  unsigned short* wrsO  = (unsigned short*)alloc((size_t)HID * 2);
  unsigned short* cbT   = (unsigned short*)alloc((size_t)KCB * LAT * 2);
  float*          cbT32 = (float*)alloc((size_t)KCB * LAT * 4);
  float*          cnorm = (float*)alloc((size_t)KCB * 4);

  float* out_zq   = (float*)d_out;                       // B*LAT (zl staged here, then z_q)
  float* out_sc   = out_zq + (size_t)B_ROWS * LAT;       // B
  float* out_rs   = out_sc + B_ROWS;                     // B
  float* out_loss = out_rs + B_ROWS;                     // 1
  float* out_ids  = out_loss + 1;                        // B (ids as float)

  // ---- prep: bf16 conversions, codebook transpose, column norms -----------
  convert_f32_bf16<<<dim3(4096), dim3(256), 0, stream>>>(z, zbf, B_ROWS * IN_DIM);
  convert_f32_bf16<<<dim3(256), dim3(256), 0, stream>>>(dec_W0, wdec0, HID * IN_DIM);
  convert_f32_bf16<<<dim3(256), dim3(256), 0, stream>>>(dec_W1, wdec1, HID * HID);
  convert_f32_bf16<<<dim3(256), dim3(256), 0, stream>>>(dec_Wout, wdecO, LAT * HID);
  convert_f32_bf16<<<dim3(256), dim3(256), 0, stream>>>(sc_W0, wsc0, HID * IN_DIM);
  convert_f32_bf16<<<dim3(256), dim3(256), 0, stream>>>(sc_W1, wsc1, HID * HID);
  convert_f32_bf16<<<dim3(4),   dim3(256), 0, stream>>>(sc_Wout, wscO, HID);
  convert_f32_bf16<<<dim3(256), dim3(256), 0, stream>>>(rs_W0, wrs0, HID * IN_DIM);
  convert_f32_bf16<<<dim3(256), dim3(256), 0, stream>>>(rs_W1, wrs1, HID * HID);
  convert_f32_bf16<<<dim3(4),   dim3(256), 0, stream>>>(rs_Wout, wrsO, HID);
  prep_codebook<<<dim3(512), dim3(256), 0, stream>>>(codebook, cbT, cbT32);
  cnorm_kernel<<<dim3(4), dim3(256), 0, stream>>>(codebook, cnorm);
  zero_kernel<<<dim3(1), dim3(32), 0, stream>>>(out_loss);

  dim3 blk(32);
  dim3 g512(B_ROWS / 32, HID / 128);  // 4096 x 4
  dim3 g256(B_ROWS / 32, LAT / 128);  // 4096 x 2

  // ---- scaler MLP ----------------------------------------------------------
  gemm_wmma_bf16<1, 0><<<g512, blk, 0, stream>>>(zbf, IN_DIM, wsc0, IN_DIM, sc_b0, hA, HID, (float*)nullptr);
  gemm_wmma_bf16<1, 0><<<g512, blk, 0, stream>>>(hA, HID, wsc1, HID, sc_b1, hB, HID, (float*)nullptr);
  head_dot<<<dim3(B_ROWS / 8), dim3(256), 0, stream>>>(hB, wscO, sc_bout, out_sc, 0);

  // ---- redshift MLP (reuse hA/hB) -----------------------------------------
  gemm_wmma_bf16<1, 0><<<g512, blk, 0, stream>>>(zbf, IN_DIM, wrs0, IN_DIM, rs_b0, hA, HID, (float*)nullptr);
  gemm_wmma_bf16<1, 0><<<g512, blk, 0, stream>>>(hA, HID, wrs1, HID, rs_b1, hB, HID, (float*)nullptr);
  head_dot<<<dim3(B_ROWS / 8), dim3(256), 0, stream>>>(hB, wrsO, rs_bout, out_rs, 1);

  // ---- decoder MLP: zl -> (zlb bf16, out_zq f32) --------------------------
  gemm_wmma_bf16<1, 0><<<g512, blk, 0, stream>>>(zbf, IN_DIM, wdec0, IN_DIM, dec_b0, hA, HID, (float*)nullptr);
  gemm_wmma_bf16<1, 0><<<g512, blk, 0, stream>>>(hA, HID, wdec1, HID, dec_b1, hB, HID, (float*)nullptr);
  gemm_wmma_bf16<0, 1><<<g256, blk, 0, stream>>>(hB, HID, wdecO, HID, dec_bout, zlb, LAT, out_zq);

  // ---- VQ: argmin over 1024 codes, gather z_q, loss, ids ------------------
  vq_argmin<<<dim3(B_ROWS / 32), blk, 0, stream>>>(zlb, cbT, cbT32, cnorm,
                                                   out_zq, out_ids, out_loss);
}
